// MultiHeadedAttention_39857296507183
// MI455X (gfx1250) — matrix operand order repaired and checked by reading, not compiled, not run on hardware
//
#include <hip/hip_runtime.h>
#include <stdint.h>

#define B_   2
#define S_   2048
#define D_   2048
#define NH_  16
#define NG_  4
#define DH_  128

typedef __attribute__((ext_vector_type(16))) __bf16 v16bf;
typedef __attribute__((ext_vector_type(8)))  __bf16 bf16x8;
typedef __attribute__((ext_vector_type(8)))  float  v8f;

union HalfPair { v16bf v; bf16x8 h[2]; };

static __device__ inline v16bf frag2(const __bf16* p0, const __bf16* p1) {
  HalfPair u;
  u.h[0] = *(const bf16x8*)p0;
  u.h[1] = *(const bf16x8*)p1;
  return u.v;
}

static __device__ inline v8f wmma_bf16(v16bf a, v16bf b, v8f c) {
  // D = A(16x32) * B(32x16) + C, f32 accumulate
  return __builtin_amdgcn_wmma_f32_16x16x32_bf16(false, a, false, b, (short)0, c,
                                                 false, false);
}

// ---- optional CDNA5 async global->LDS path (ASYNCcnt) ----------------------
#if defined(__has_builtin)
#if __has_builtin(__builtin_amdgcn_global_load_async_to_lds_b128) && \
    __has_builtin(__builtin_amdgcn_s_wait_asynccnt)
#define HAVE_ASYNC_LDS 1
#endif
#endif

#ifdef HAVE_ASYNC_LDS
// builtin expects pointers to a 4 x i32 vector in AS(1)/AS(3)
typedef int vint4_t __attribute__((vector_size(4 * sizeof(int))));
static __device__ inline void async_copy_b128(const void* gsrc, void* ldst) {
  // generic LDS pointers carry the LDS offset in their low 32 bits
  __builtin_amdgcn_global_load_async_to_lds_b128(
      (__attribute__((address_space(1))) vint4_t*)(uintptr_t)gsrc,
      (__attribute__((address_space(3))) vint4_t*)(uint32_t)(uintptr_t)ldst,
      0, 0);
}
#endif

// ---------------------------------------------------------------------------
// Kernel 1: generic projection GEMM (double-buffered LDS, register staging).
//   C[m, n] = sum_k X[m, k] * W[n, k] + bias[n]   (X fp32 [M, D_], W fp32 [N, D_])
// Output written as bf16 into [b, head, s, dh] layout (head = n/DH_, dh = n%DH_).
// Block = 256 threads = 8 waves (4 over M x 2 over N). Tile 128x128, BK=32.
// ---------------------------------------------------------------------------
__global__ __launch_bounds__(256)
void proj_kernel(const float* __restrict__ X, const float* __restrict__ W,
                 const float* __restrict__ bias, __bf16* __restrict__ out,
                 int heads) {
  __shared__ __bf16 As[2][128][40];
  __shared__ __bf16 Bs[2][128][40];

  const int t = threadIdx.x;
  const int lane = t & 31;
  const int wave = t >> 5;
  const int waveM = wave & 3;   // 4 waves along M (32 rows each)
  const int waveN = wave >> 2;  // 2 waves along N (64 cols each)
  const int mBase = blockIdx.y * 128;
  const int nBase = blockIdx.x * 128;
  const int mrow = lane & 15;
  const int khalf = lane >> 4;

  v8f acc[2][4];
#pragma unroll
  for (int i = 0; i < 2; ++i)
#pragma unroll
    for (int j = 0; j < 4; ++j) acc[i][j] = {};

  float4 fa[4], fb[4];
  auto load_regs = [&](int k0) {
#pragma unroll
    for (int i = 0; i < 4; ++i) {
      int idx = t + i * 256;  // 0..1023
      int r = idx >> 3, c4 = (idx & 7) << 2;
      fa[i] = *(const float4*)&X[(size_t)(mBase + r) * D_ + k0 + c4];
      fb[i] = *(const float4*)&W[(size_t)(nBase + r) * D_ + k0 + c4];
    }
  };
  auto store_lds = [&](int buf) {
#pragma unroll
    for (int i = 0; i < 4; ++i) {
      int idx = t + i * 256;
      int r = idx >> 3, c4 = (idx & 7) << 2;
      As[buf][r][c4 + 0] = (__bf16)fa[i].x; As[buf][r][c4 + 1] = (__bf16)fa[i].y;
      As[buf][r][c4 + 2] = (__bf16)fa[i].z; As[buf][r][c4 + 3] = (__bf16)fa[i].w;
      Bs[buf][r][c4 + 0] = (__bf16)fb[i].x; Bs[buf][r][c4 + 1] = (__bf16)fb[i].y;
      Bs[buf][r][c4 + 2] = (__bf16)fb[i].z; Bs[buf][r][c4 + 3] = (__bf16)fb[i].w;
    }
  };

  load_regs(0);
  store_lds(0);
  __syncthreads();

  int buf = 0;
  for (int k0 = 0; k0 < D_; k0 += 32) {
    const bool hasNext = (k0 + 32 < D_);
    if (hasNext) load_regs(k0 + 32);  // global loads overlap with WMMA below

    v16bf a[2], b[4];
#pragma unroll
    for (int i = 0; i < 2; ++i) {
      int m = waveM * 32 + i * 16 + mrow;
      a[i] = frag2(&As[buf][m][khalf * 8], &As[buf][m][16 + khalf * 8]);
    }
#pragma unroll
    for (int j = 0; j < 4; ++j) {
      int n = waveN * 64 + j * 16 + mrow;
      b[j] = frag2(&Bs[buf][n][khalf * 8], &Bs[buf][n][16 + khalf * 8]);
    }
#pragma unroll
    for (int i = 0; i < 2; ++i)
#pragma unroll
      for (int j = 0; j < 4; ++j) acc[i][j] = wmma_bf16(a[i], b[j], acc[i][j]);

    if (hasNext) store_lds(buf ^ 1);  // fill other buffer; current reads done
    __syncthreads();
    buf ^= 1;
  }

  // epilogue: + bias, write bf16 to [b, head, s, dh]
#pragma unroll
  for (int j = 0; j < 4; ++j) {
    int n = nBase + waveN * 64 + j * 16 + mrow;
    float bv = bias[n];
    int head = n >> 7;  // n / DH_
    int dh = n & 127;
#pragma unroll
    for (int i = 0; i < 2; ++i) {
#pragma unroll
      for (int r = 0; r < 8; ++r) {
        int m = mBase + waveM * 32 + i * 16 + r + 8 * khalf;
        int b = m >> 11;   // m / S_
        int s = m & 2047;  // m % S_
        out[(((size_t)b * heads + head) * S_ + s) * DH_ + dh] =
            (__bf16)(acc[i][j][r] + bv);
      }
    }
  }
}

// ---------------------------------------------------------------------------
// Kernel 2: flash attention per (b, head, 64-row q tile).
// 128 threads = 4 waves, each wave owns 16 q rows. KV tiles of 64.
// Q/K tiles copied via async global->LDS when available.
// ---------------------------------------------------------------------------
__global__ __launch_bounds__(128)
void flash_kernel(const __bf16* __restrict__ q_ws, const __bf16* __restrict__ k_ws,
                  const __bf16* __restrict__ v_ws, __bf16* __restrict__ ao_ws) {
  __shared__ __bf16 Qs[64][136];
  __shared__ __bf16 Ks[64][136];
  __shared__ __bf16 Vt[128][72];    // V transposed: [dh][sk]
  __shared__ __bf16 Ps[4][16][72];  // per-wave P staging

  const int t = threadIdx.x;
  const int lane = t & 31;
  const int wave = t >> 5;
  const int mrow = lane & 15;
  const int khalf = lane >> 4;

  const int bh = blockIdx.y;  // b*16 + head
  const int b = bh >> 4;
  const int head = bh & 15;
  const int g = head >> 2;

  const __bf16* Qg = q_ws + ((size_t)bh * S_ + blockIdx.x * 64) * DH_;
  const __bf16* Kg = k_ws + (size_t)(b * NG_ + g) * S_ * DH_;
  const __bf16* Vg = v_ws + (size_t)(b * NG_ + g) * S_ * DH_;

  // load Q tile (64 x 128 bf16) into LDS
#pragma unroll
  for (int i = 0; i < 8; ++i) {
    int idx = t + i * 128;  // 0..1023
    int r = idx >> 4;
    int c8 = (idx & 15) << 3;
#ifdef HAVE_ASYNC_LDS
    async_copy_b128(&Qg[(size_t)r * DH_ + c8], &Qs[r][c8]);
#else
    *(bf16x8*)&Qs[r][c8] = *(const bf16x8*)&Qg[(size_t)r * DH_ + c8];
#endif
  }
#ifdef HAVE_ASYNC_LDS
  __builtin_amdgcn_s_wait_asynccnt(0);
#endif
  __syncthreads();

  // Q A-fragments (16 rows x 128 dh -> 4 frags of K=32)
  v16bf qf[4];
#pragma unroll
  for (int ks = 0; ks < 4; ++ks) {
    int m = wave * 16 + mrow;
    qf[ks] = frag2(&Qs[m][ks * 32 + khalf * 8], &Qs[m][ks * 32 + 16 + khalf * 8]);
  }

  v8f o[8];
#pragma unroll
  for (int f = 0; f < 8; ++f) o[f] = {};
  float mprev[8], lrow[8];
#pragma unroll
  for (int r = 0; r < 8; ++r) { mprev[r] = -3.0e38f; lrow[r] = 0.0f; }

  const float scale = 0.088388347648318447f;  // 1/sqrt(128)

  for (int kv = 0; kv < S_ / 64; ++kv) {
    __syncthreads();  // protect Ks/Vt reuse from previous iteration
    // K tile: async raw copy; V tile: register transpose
#pragma unroll
    for (int i = 0; i < 8; ++i) {
      int idx = t + i * 128;
      int r = idx >> 4;
      int c8 = (idx & 15) << 3;
#ifdef HAVE_ASYNC_LDS
      async_copy_b128(&Kg[(size_t)(kv * 64 + r) * DH_ + c8], &Ks[r][c8]);
#else
      *(bf16x8*)&Ks[r][c8] = *(const bf16x8*)&Kg[(size_t)(kv * 64 + r) * DH_ + c8];
#endif
      bf16x8 vv = *(const bf16x8*)&Vg[(size_t)(kv * 64 + r) * DH_ + c8];
#pragma unroll
      for (int e = 0; e < 8; ++e) Vt[c8 + e][r] = vv[e];
    }
    if (kv + 1 < S_ / 64) {  // prefetch next KV tile -> global_prefetch_b8
      const __bf16* pk = &Kg[(size_t)((kv + 1) * 64 + (t >> 1)) * DH_ + (t & 1) * 64];
      const __bf16* pv = &Vg[(size_t)((kv + 1) * 64 + (t >> 1)) * DH_ + (t & 1) * 64];
      __builtin_prefetch(pk, 0, 3);
      __builtin_prefetch(pv, 0, 3);
    }
#ifdef HAVE_ASYNC_LDS
    __builtin_amdgcn_s_wait_asynccnt(0);
#endif
    __syncthreads();

    // scores: S = Q * K^T  (16 x 64 per wave)
    v8f sc[4];
#pragma unroll
    for (int nt = 0; nt < 4; ++nt) {
      sc[nt] = {};
#pragma unroll
      for (int ks = 0; ks < 4; ++ks) {
        int n = nt * 16 + mrow;  // column = key row
        v16bf kf = frag2(&Ks[n][ks * 32 + khalf * 8],
                         &Ks[n][ks * 32 + 16 + khalf * 8]);
        sc[nt] = wmma_bf16(qf[ks], kf, sc[nt]);
      }
#pragma unroll
      for (int r = 0; r < 8; ++r) sc[nt][r] *= scale;
    }

    // online softmax (rows live in C-frag layout: row = r + 8*khalf)
    float mnew[8], alpha[8], psum[8];
#pragma unroll
    for (int r = 0; r < 8; ++r) {
      float mx = fmaxf(fmaxf(sc[0][r], sc[1][r]), fmaxf(sc[2][r], sc[3][r]));
#pragma unroll
      for (int sh = 1; sh < 16; sh <<= 1) mx = fmaxf(mx, __shfl_xor(mx, sh, 32));
      mnew[r] = fmaxf(mx, mprev[r]);
      alpha[r] = __expf(mprev[r] - mnew[r]);
      psum[r] = 0.0f;
    }
#pragma unroll
    for (int nt = 0; nt < 4; ++nt)
#pragma unroll
      for (int r = 0; r < 8; ++r) {
        float p = __expf(sc[nt][r] - mnew[r]);
        sc[nt][r] = p;
        psum[r] += p;
      }
#pragma unroll
    for (int r = 0; r < 8; ++r) {
#pragma unroll
      for (int sh = 1; sh < 16; sh <<= 1) psum[r] += __shfl_xor(psum[r], sh, 32);
      lrow[r] = lrow[r] * alpha[r] + psum[r];
      mprev[r] = mnew[r];
    }
    // rescale existing output accumulators
#pragma unroll
    for (int f = 0; f < 8; ++f)
#pragma unroll
      for (int r = 0; r < 8; ++r) o[f][r] *= alpha[r];

    // stage P (bf16) into LDS row-major so it can be re-read as an A-frag
#pragma unroll
    for (int nt = 0; nt < 4; ++nt)
#pragma unroll
      for (int r = 0; r < 8; ++r)
        Ps[wave][r + 8 * khalf][nt * 16 + mrow] = (__bf16)sc[nt][r];
    __syncthreads();

    // O += P * V   (A = P 16x64, B = V 64x128 via transposed LDS)
#pragma unroll
    for (int nt2 = 0; nt2 < 8; ++nt2) {
#pragma unroll
      for (int ks2 = 0; ks2 < 2; ++ks2) {
        v16bf pf = frag2(&Ps[wave][mrow][ks2 * 32 + khalf * 8],
                         &Ps[wave][mrow][ks2 * 32 + 16 + khalf * 8]);
        int n = nt2 * 16 + mrow;  // dh column
        v16bf vf = frag2(&Vt[n][ks2 * 32 + khalf * 8],
                         &Vt[n][ks2 * 32 + 16 + khalf * 8]);
        o[nt2] = wmma_bf16(pf, vf, o[nt2]);
      }
    }
  }

  // epilogue: normalize and write bf16 [b, head, s, dh]
  __bf16* Og = ao_ws + ((size_t)bh * S_ + blockIdx.x * 64) * DH_;
#pragma unroll
  for (int nt2 = 0; nt2 < 8; ++nt2) {
    int dh = nt2 * 16 + mrow;
#pragma unroll
    for (int r = 0; r < 8; ++r) {
      int srow = wave * 16 + r + 8 * khalf;
      Og[(size_t)srow * DH_ + dh] = (__bf16)(o[nt2][r] / lrow[r]);
    }
  }
}

// ---------------------------------------------------------------------------
// Kernel 3: output projection (double-buffered).
//   out[m, n] = sum_e AO[m, e] * WO[n, e] + bO[n], m = b*S+s, e head-permuted.
// ---------------------------------------------------------------------------
__global__ __launch_bounds__(256)
void outproj_kernel(const __bf16* __restrict__ AO, const float* __restrict__ WO,
                    const float* __restrict__ bO, float* __restrict__ out) {
  __shared__ __bf16 As[2][128][40];
  __shared__ __bf16 Bs[2][128][40];

  const int t = threadIdx.x;
  const int lane = t & 31;
  const int wave = t >> 5;
  const int waveM = wave & 3;
  const int waveN = wave >> 2;
  const int mBase = blockIdx.y * 128;
  const int nBase = blockIdx.x * 128;
  const int mrow = lane & 15;
  const int khalf = lane >> 4;

  v8f acc[2][4];
#pragma unroll
  for (int i = 0; i < 2; ++i)
#pragma unroll
    for (int j = 0; j < 4; ++j) acc[i][j] = {};

  bf16x8 ga[2];
  float4 gb[4];
  auto load_regs = [&](int k0) {
#pragma unroll
    for (int i = 0; i < 2; ++i) {  // A: bf16, head-permuted k index
      int idx = t + i * 256;       // 0..511
      int r = idx >> 2, c8 = (idx & 3) << 3;
      int m = mBase + r;
      int b = m >> 11, s = m & 2047;
      int e = k0 + c8;
      int head = e >> 7, dh = e & 127;
      ga[i] = *(const bf16x8*)&AO[(((size_t)b * NH_ + head) * S_ + s) * DH_ + dh];
    }
#pragma unroll
    for (int i = 0; i < 4; ++i) {  // W: fp32
      int idx = t + i * 256;
      int r = idx >> 3, c4 = (idx & 7) << 2;
      gb[i] = *(const float4*)&WO[(size_t)(nBase + r) * D_ + k0 + c4];
    }
  };
  auto store_lds = [&](int buf) {
#pragma unroll
    for (int i = 0; i < 2; ++i) {
      int idx = t + i * 256;
      int r = idx >> 2, c8 = (idx & 3) << 3;
      *(bf16x8*)&As[buf][r][c8] = ga[i];
    }
#pragma unroll
    for (int i = 0; i < 4; ++i) {
      int idx = t + i * 256;
      int r = idx >> 3, c4 = (idx & 7) << 2;
      Bs[buf][r][c4 + 0] = (__bf16)gb[i].x; Bs[buf][r][c4 + 1] = (__bf16)gb[i].y;
      Bs[buf][r][c4 + 2] = (__bf16)gb[i].z; Bs[buf][r][c4 + 3] = (__bf16)gb[i].w;
    }
  };

  load_regs(0);
  store_lds(0);
  __syncthreads();

  int buf = 0;
  for (int k0 = 0; k0 < D_; k0 += 32) {
    const bool hasNext = (k0 + 32 < D_);
    if (hasNext) load_regs(k0 + 32);

    v16bf a[2], b[4];
#pragma unroll
    for (int i = 0; i < 2; ++i) {
      int m = waveM * 32 + i * 16 + mrow;
      a[i] = frag2(&As[buf][m][khalf * 8], &As[buf][m][16 + khalf * 8]);
    }
#pragma unroll
    for (int j = 0; j < 4; ++j) {
      int n = waveN * 64 + j * 16 + mrow;
      b[j] = frag2(&Bs[buf][n][khalf * 8], &Bs[buf][n][16 + khalf * 8]);
    }
#pragma unroll
    for (int i = 0; i < 2; ++i)
#pragma unroll
      for (int j = 0; j < 4; ++j) acc[i][j] = wmma_bf16(a[i], b[j], acc[i][j]);

    if (hasNext) store_lds(buf ^ 1);
    __syncthreads();
    buf ^= 1;
  }

#pragma unroll
  for (int j = 0; j < 4; ++j) {
    int n = nBase + waveN * 64 + j * 16 + mrow;
    float bv = bO[n];
#pragma unroll
    for (int i = 0; i < 2; ++i) {
#pragma unroll
      for (int r = 0; r < 8; ++r) {
        int m = mBase + waveM * 32 + i * 16 + r + 8 * khalf;
        out[(size_t)m * D_ + n] = acc[i][j][r] + bv;
      }
    }
  }
}

// ---------------------------------------------------------------------------
extern "C" void kernel_launch(void* const* d_in, const int* in_sizes, int n_in,
                              void* d_out, int out_size, void* d_ws, size_t ws_size,
                              hipStream_t stream) {
  (void)in_sizes; (void)n_in; (void)out_size; (void)ws_size;
  const float* query = (const float*)d_in[0];
  const float* key   = (const float*)d_in[1];
  const float* value = (const float*)d_in[2];
  const float* WQ    = (const float*)d_in[3];  // [16*128, 2048] flattened
  const float* bQ    = (const float*)d_in[4];
  const float* WK    = (const float*)d_in[5];  // [4*128, 2048]
  const float* bK    = (const float*)d_in[6];
  const float* WV    = (const float*)d_in[7];
  const float* bV    = (const float*)d_in[8];
  const float* WO    = (const float*)d_in[9];  // [2048, 2048]
  const float* bO    = (const float*)d_in[10];

  __bf16* q_ws  = (__bf16*)d_ws;                       // [B,16,S,128]
  __bf16* k_ws  = q_ws + (size_t)B_ * NH_ * S_ * DH_;  // [B, 4,S,128]
  __bf16* v_ws  = k_ws + (size_t)B_ * NG_ * S_ * DH_;
  __bf16* ao_ws = v_ws + (size_t)B_ * NG_ * S_ * DH_;  // [B,16,S,128]

  const int M = B_ * S_;  // 4096
  proj_kernel<<<dim3(2048 / 128, M / 128), 256, 0, stream>>>(query, WQ, bQ, q_ws, NH_);
  proj_kernel<<<dim3(512 / 128, M / 128), 256, 0, stream>>>(key, WK, bK, k_ws, NG_);
  proj_kernel<<<dim3(512 / 128, M / 128), 256, 0, stream>>>(value, WV, bV, v_ws, NG_);
  flash_kernel<<<dim3(S_ / 64, B_ * NH_), 128, 0, stream>>>(q_ws, k_ws, v_ws, ao_ws);
  outproj_kernel<<<dim3(D_ / 128, M / 128), 256, 0, stream>>>(ao_ws, WO, bO,
                                                              (float*)d_out);
}
